// CGMM_9732395893089
// MI455X (gfx1250) — matrix operand order repaired and checked by reading, not compiled, not run on hardware
//
#include <hip/hip_runtime.h>
#include <hip/hip_bf16.h>

typedef __attribute__((ext_vector_type(16))) _Float16 v16h;
typedef __attribute__((ext_vector_type(8)))  float    v8f;

#define CC 8
#define GG 8
#define MM 32

// ---------------------------------------------------------------------------
// Kernel 1: precompute all parameter tables (single block, tiny work).
//   T0 [x][g][c]  = sm_Pi[c,g] * sm_B0[c,x,g]        (layer-0 numerator)
//   LT0[x][g][c]  = log(T0)
//   E1 [x][g][c]  = sm_B1[c,x,g]                      (layer-1 emission)
//   LE1[x][g][c]  = log(E1)
//   LQ [g][h][c]  = log(sm_Q[c,h,g])                  (log transition)
//   QH [g][c][k]  = (f16) sm_Q[c,k,g]                 (WMMA B-matrix source)
// ---------------------------------------------------------------------------
__global__ void k_params(const float* __restrict__ B0, const float* __restrict__ Pi,
                         const float* __restrict__ Q,  const float* __restrict__ B1,
                         float* __restrict__ T0, float* __restrict__ LT0,
                         float* __restrict__ E1, float* __restrict__ LE1,
                         float* __restrict__ LQ, _Float16* __restrict__ QH) {
  __shared__ float sB0[CC * MM * GG];
  __shared__ float sB1[CC * MM * GG];
  __shared__ float sPi[CC * GG];
  __shared__ float sQ[CC * CC * GG];
  const int tid = threadIdx.x;

  if (tid < 64) {
    const int c = tid >> 3, g = tid & 7;
    // softmax over M for B0[c,:,g]
    {
      float mx = -1e30f;
      for (int m = 0; m < MM; ++m) mx = fmaxf(mx, B0[c * MM * GG + m * GG + g]);
      float s = 0.f;
      for (int m = 0; m < MM; ++m) s += expf(B0[c * MM * GG + m * GG + g] - mx);
      const float inv = 1.f / s;
      for (int m = 0; m < MM; ++m)
        sB0[c * MM * GG + m * GG + g] = expf(B0[c * MM * GG + m * GG + g] - mx) * inv;
    }
    // softmax over M for B1[c,:,g]
    {
      float mx = -1e30f;
      for (int m = 0; m < MM; ++m) mx = fmaxf(mx, B1[c * MM * GG + m * GG + g]);
      float s = 0.f;
      for (int m = 0; m < MM; ++m) s += expf(B1[c * MM * GG + m * GG + g] - mx);
      const float inv = 1.f / s;
      for (int m = 0; m < MM; ++m)
        sB1[c * MM * GG + m * GG + g] = expf(B1[c * MM * GG + m * GG + g] - mx) * inv;
    }
    // softmax over new-state axis (axis 0) of Q[:,k,g] for this (k,g)
    {
      const int k = c;  // reuse tid decomposition as (k,g)
      float mx = -1e30f;
      for (int cc = 0; cc < CC; ++cc) mx = fmaxf(mx, Q[cc * 64 + k * 8 + g]);
      float s = 0.f;
      for (int cc = 0; cc < CC; ++cc) s += expf(Q[cc * 64 + k * 8 + g] - mx);
      const float inv = 1.f / s;
      for (int cc = 0; cc < CC; ++cc)
        sQ[cc * 64 + k * 8 + g] = expf(Q[cc * 64 + k * 8 + g] - mx) * inv;
    }
  }
  if (tid < 8) {  // softmax over C for Pi[:,g]
    const int g = tid;
    float mx = -1e30f;
    for (int c = 0; c < CC; ++c) mx = fmaxf(mx, Pi[c * 8 + g]);
    float s = 0.f;
    for (int c = 0; c < CC; ++c) s += expf(Pi[c * 8 + g] - mx);
    const float inv = 1.f / s;
    for (int c = 0; c < CC; ++c) sPi[c * 8 + g] = expf(Pi[c * 8 + g] - mx) * inv;
  }
  __syncthreads();

  for (int idx = tid; idx < MM * GG * CC; idx += blockDim.x) {
    const int xm = idx >> 6, g = (idx >> 3) & 7, c = idx & 7;
    const float v0 = sPi[c * 8 + g] * sB0[c * MM * GG + xm * GG + g];
    T0[idx]  = v0;
    LT0[idx] = logf(v0);
    const float v1 = sB1[c * MM * GG + xm * GG + g];
    E1[idx]  = v1;
    LE1[idx] = logf(v1);
  }
  for (int idx = tid; idx < GG * CC * CC; idx += blockDim.x) {
    const int g = idx >> 6, h = (idx >> 3) & 7, c = idx & 7;
    const float v = sQ[c * 64 + h * 8 + g];
    LQ[idx] = logf(v);
    QH[g * 64 + c * 8 + h] = (_Float16)v;   // [g][c][k] layout for WMMA B
  }
}

// ---------------------------------------------------------------------------
// Kernel 2: layer 0 per node: posterior, lik0 accumulation, argmax -> packed h0
// ---------------------------------------------------------------------------
__global__ void k_layer0(const int* __restrict__ x, const int* __restrict__ batch,
                         const float* __restrict__ T0, const float* __restrict__ LT0,
                         unsigned* __restrict__ h0p, float* __restrict__ out, int N) {
  const int n = blockIdx.x * blockDim.x + threadIdx.x;
  if (n >= N) return;
  const int xn = x[n];
  const int b  = batch[n];
  const float* t0 = T0  + xn * 64;
  const float* lt = LT0 + xn * 64;
  unsigned hp = 0;
  #pragma unroll
  for (int g = 0; g < GG; ++g) {
    float v[8];
    float s = 0.f;
    #pragma unroll
    for (int c = 0; c < CC; ++c) { v[c] = t0[g * 8 + c]; s += v[c]; }
    const float invs = 1.f / s;
    float best = v[0];
    int arg = 0;
    float lik = v[0] * invs * lt[g * 8];
    #pragma unroll
    for (int c = 1; c < CC; ++c) {
      lik += v[c] * invs * lt[g * 8 + c];
      if (v[c] > best) { best = v[c]; arg = c; }   // first-max like jnp.argmax
    }
    hp |= (unsigned)arg << (4 * g);
    atomicAdd(&out[b * 16 + g], -lik);
  }
  h0p[n] = hp;
}

// ---------------------------------------------------------------------------
// Kernel 3: edge histogram. hist is [n][g][k/2] u32 with two u16 counters.
//   hist[n,g,k] = #edges with src=n and h0[dst,g]=k
// ---------------------------------------------------------------------------
__global__ void k_edges(const int* __restrict__ ei, const unsigned* __restrict__ h0p,
                        unsigned* __restrict__ hist, int E) {
  const int e = blockIdx.x * blockDim.x + threadIdx.x;
  if (e >= E) return;
  const int src = ei[e];
  const int dst = ei[E + e];
  const unsigned hp = h0p[dst];
  const size_t base = (size_t)src * 32u;
  #pragma unroll
  for (int g = 0; g < GG; ++g) {
    const unsigned k = (hp >> (4 * g)) & 15u;
    atomicAdd(&hist[base + g * 4 + (k >> 1)], 1u << (16 * (k & 1)));
  }
}

// ---------------------------------------------------------------------------
// Kernel 4: layer 1. One wave = one 16-node tile.
//   Qu numerators via V_WMMA_F32_16X16X32_F16, two g-components per WMMA
//   (block-diagonal packing: K 0..7 -> g0 rows, K 8..15 -> g1 rows,
//    N 0..7 -> g0 states, N 8..15 -> g1 states, K 16..31 zero).
//   ISA layouts used: 16-bit A: lanes 0-15 hold K 0..7 / lanes 16-31 hold
//   K 8..15 in elements 0..7; 16-bit B: element e == K=e (lanes 0-15);
//   f32 D: lane%16 = N, vgpr r -> M = r + 8*(lane>=16).
//   All vector element indices are compile-time constants (no scratch spill);
//   lane-dependent placement is done with selects, not dynamic indexing.
// ---------------------------------------------------------------------------
__global__ __launch_bounds__(32)
void k_layer1(const int* __restrict__ x, const int* __restrict__ batch,
              const unsigned* __restrict__ h0p, const unsigned* __restrict__ hist,
              const float* __restrict__ E1, const float* __restrict__ LE1,
              const float* __restrict__ LQ, const _Float16* __restrict__ QH,
              float* __restrict__ out, int N) {
  __shared__ float qu[16][GG][CC];   // [node_local][g][c] Qu numerators
  const int lane = threadIdx.x;
  const int nb = blockIdx.x * 16;
  if (nb >= N) return;

  const _Float16 h0c = (_Float16)0.0f;
  const bool loB = (lane < 8);                 // this lane feeds B rows K=0..7 (g0)
  const bool hiB = (lane >= 24);               // this lane feeds B rows K=8..15 (g1)

  #pragma unroll
  for (int t = 0; t < 4; ++t) {
    // ---- B matrix (sm_Q in f16, block-diagonal over the g pair) ----
    // Every lane loads a (clamped) valid sm_Q row; placement via selects only.
    const int gB = 2 * t + ((lane >> 3) & 1);
    const int cB = lane & 7;
    const _Float16* qp = QH + gB * 64 + cB * 8;
    v16h bm;
    #pragma unroll
    for (int k = 0; k < 8; ++k) {
      const _Float16 v = qp[k];
      bm[k]     = (loB || hiB) ? v : h0c;
      bm[k + 8] = h0c;
    }
    // ---- A matrix (histogram counts, exact in f16) ----
    const int nl = lane & 15;
    int node = nb + nl; if (node >= N) node = N - 1;
    const int gsel = 2 * t + (lane >> 4);
    const uint4 w4 = *(const uint4*)(hist + (size_t)node * 32 + gsel * 4);
    v16h am;
    am[0] = (_Float16)(float)(w4.x & 0xFFFFu);
    am[1] = (_Float16)(float)(w4.x >> 16);
    am[2] = (_Float16)(float)(w4.y & 0xFFFFu);
    am[3] = (_Float16)(float)(w4.y >> 16);
    am[4] = (_Float16)(float)(w4.z & 0xFFFFu);
    am[5] = (_Float16)(float)(w4.z >> 16);
    am[6] = (_Float16)(float)(w4.w & 0xFFFFu);
    am[7] = (_Float16)(float)(w4.w >> 16);
    #pragma unroll
    for (int i = 8; i < 16; ++i) am[i] = h0c;   // K=16..31 unused -> zero
    v8f dm;
    #pragma unroll
    for (int i = 0; i < 8; ++i) dm[i] = 0.0f;
    dm = __builtin_amdgcn_wmma_f32_16x16x32_f16(false, am, false, bm,
                                                (short)0, dm, false, false);
    // ---- scatter D into [node][g][c] LDS layout ----
    const int Ncol  = lane & 15;
    const int g     = 2 * t + (Ncol >> 3);
    const int c     = Ncol & 7;
    const int mbase = (lane < 16) ? 0 : 8;
    #pragma unroll
    for (int r = 0; r < 8; ++r) qu[mbase + r][g][c] = dm[r];
  }
  __syncthreads();

  // ---- per-(node,g) posterior + likelihood: 128 pairs over 32 lanes ----
  #pragma unroll
  for (int it = 0; it < 4; ++it) {
    const int pair = lane + 32 * it;
    const int nl = pair >> 3;
    const int g  = pair & 7;
    const int n  = nb + nl;
    if (n >= N) continue;
    // degree = sum of this node's histogram row (identical for every g)
    const uint4 w4 = *(const uint4*)(hist + (size_t)n * 32 + g * 4);
    const unsigned cnt = (w4.x & 0xFFFFu) + (w4.x >> 16) +
                         (w4.y & 0xFFFFu) + (w4.y >> 16) +
                         (w4.z & 0xFFFFu) + (w4.z >> 16) +
                         (w4.w & 0xFFFFu) + (w4.w >> 16);
    const float invc = 1.0f / fmaxf((float)cnt, 1.0f);
    const int xn = x[n];
    const int h  = (h0p[n] >> (4 * g)) & 15;
    const float* e1 = E1  + xn * 64 + g * 8;
    const float* le = LE1 + xn * 64 + g * 8;
    const float* lq = LQ  + g * 64 + h * 8;
    float pu[8];
    float s = 0.0f;
    #pragma unroll
    for (int c = 0; c < CC; ++c) {
      const float q = qu[nl][g][c] * invc;   // Qu = summ / max(cnt,1)
      pu[c] = q * e1[c];
      s += pu[c];
    }
    const float invs = 1.0f / (s + 1e-12f);
    float lik = 0.0f;
    #pragma unroll
    for (int c = 0; c < CC; ++c) lik += pu[c] * invs * (lq[c] + le[c]);
    atomicAdd(&out[batch[n] * 16 + 8 + g], -lik);
  }
}

// ---------------------------------------------------------------------------
extern "C" void kernel_launch(void* const* d_in, const int* in_sizes, int n_in,
                              void* d_out, int out_size, void* d_ws, size_t ws_size,
                              hipStream_t stream) {
  const int*   x   = (const int*)d_in[0];
  const int*   ei  = (const int*)d_in[1];
  const int*   bat = (const int*)d_in[2];
  const float* B0  = (const float*)d_in[3];
  const float* Pi  = (const float*)d_in[4];
  const float* Q   = (const float*)d_in[5];
  const float* B1  = (const float*)d_in[6];
  float* out = (float*)d_out;
  const int N = in_sizes[0];
  const int E = in_sizes[1] / 2;

  // workspace layout (bytes): tables | packed h0 | packed histogram
  char* ws = (char*)d_ws;
  float*    T0  = (float*)(ws + 0);        // 2048 f32
  float*    LT0 = (float*)(ws + 8192);
  float*    E1  = (float*)(ws + 16384);
  float*    LE1 = (float*)(ws + 24576);
  float*    LQ  = (float*)(ws + 32768);    // 512 f32
  _Float16* QH  = (_Float16*)(ws + 34816); // 512 f16
  unsigned* h0p = (unsigned*)(ws + 35840); // N u32
  const size_t histOff = 35840 + (((size_t)N * 4 + 255) / 256) * 256;
  unsigned* hist = (unsigned*)(ws + histOff); // N*32 u32 (N*8g*8k u16)

  hipMemsetAsync(out, 0, (size_t)out_size * sizeof(float), stream);
  hipMemsetAsync(hist, 0, (size_t)N * 32 * sizeof(unsigned), stream);

  k_params<<<1, 256, 0, stream>>>(B0, Pi, Q, B1, T0, LT0, E1, LE1, LQ, QH);
  k_layer0<<<(N + 255) / 256, 256, 0, stream>>>(x, bat, T0, LT0, h0p, out, N);
  k_edges<<<(E + 255) / 256, 256, 0, stream>>>(ei, h0p, hist, E);
  k_layer1<<<(N + 15) / 16, 32, 0, stream>>>(x, bat, h0p, hist, E1, LE1, LQ, QH, out, N);
}
